// AntiSymmetricDGN_14353780703435
// MI455X (gfx1250) — matrix operand order repaired and checked by reading, not compiled, not run on hardware
//
#include <hip/hip_runtime.h>
#include <math.h>

typedef _Float16 v16h __attribute__((ext_vector_type(16)));
typedef float    v8f  __attribute__((ext_vector_type(8)));
typedef _Float16 h2v  __attribute__((ext_vector_type(2)));

#define GN 50000
#define GE 600000
#define GIN 256
#define GH 128
#define GOUT 40
#define EPSV 0.1f
#define GAMMAV 0.1f
#define LSLOPE 0.01f

// ---------------- prep kernels (f32 weights -> f16 Bt[n][k] layout) ----------------

__global__ void cvt_copy_k(const float* __restrict__ src, _Float16* __restrict__ dst, int n) {
  int i = blockIdx.x * blockDim.x + threadIdx.x;
  if (i < n) dst[i] = (_Float16)src[i];
}

// Bt[n*K + k] = src[k*N + n]   (for p = h @ W, W row-major [K][N])
__global__ void cvt_transpose_k(const float* __restrict__ src, _Float16* __restrict__ dst, int K, int N) {
  int i = blockIdx.x * blockDim.x + threadIdx.x;
  if (i < N * K) {
    int n = i / K, k = i % K;
    dst[i] = (_Float16)src[k * N + n];
  }
}

// Bt[j*d + k] = W[j,k] - W[k,j] - gamma*(j==k)   (aW for h @ aW^T)
__global__ void antisym_prep_k(const float* __restrict__ W, _Float16* __restrict__ dst, int d, float gamma) {
  int i = blockIdx.x * blockDim.x + threadIdx.x;
  if (i < d * d) {
    int j = i / d, k = i % d;
    float v = W[j * d + k] - W[k * d + j] - ((j == k) ? gamma : 0.f);
    dst[i] = (_Float16)v;
  }
}

// pad w_fc rows 40 -> 48
__global__ void wfc_prep_k(const float* __restrict__ wfc, _Float16* __restrict__ dst, int K, int Nout, int Npad) {
  int i = blockIdx.x * blockDim.x + threadIdx.x;
  if (i < Npad * K) {
    int j = i / K, k = i % K;
    dst[i] = (j < Nout) ? (_Float16)wfc[j * K + k] : (_Float16)0.f;
  }
}

// ---------------- degree / norm ----------------

__global__ void deg_init_k(float* __restrict__ deg, int n) {
  int i = blockIdx.x * blockDim.x + threadIdx.x;
  if (i < n) deg[i] = 1.0f;  // self loop
}

__global__ void deg_acc_k(const int* __restrict__ ei, float* __restrict__ deg, int E, int N) {
  int e = blockIdx.x * blockDim.x + threadIdx.x;
  if (e < E) {
    int d = ei[E + e];
    if ((unsigned)d < (unsigned)N) atomicAdd(&deg[d], 1.0f);
  }
}

__global__ void dinv_k(const float* __restrict__ deg, float* __restrict__ dinv, int n) {
  int i = blockIdx.x * blockDim.x + threadIdx.x;
  if (i < n) {
    float d = deg[i];
    dinv[i] = (d > 0.f) ? rsqrtf(d) : 0.f;
  }
}

// ---------------- GCN aggregation ----------------

// agg[n,:] = dinv[n]^2 * p[n,:]  (self-loop term, also zero-inits agg)
__global__ void agg_init_k(const float* __restrict__ p, const float* __restrict__ dinv,
                           float* __restrict__ agg, int N, int H) {
  int i = blockIdx.x * blockDim.x + threadIdx.x;
  if (i < N * H) {
    int n = i / H;
    float w = dinv[n] * dinv[n];
    agg[i] = w * p[i];
  }
}

// agg[dst,:] += dinv[src]*dinv[dst] * p[src,:], 4 floats per thread
__global__ void scatter_k(const int* __restrict__ ei, const float* __restrict__ p,
                          const float* __restrict__ dinv, float* __restrict__ agg,
                          int E, int H, int N) {
  int idx = blockIdx.x * blockDim.x + threadIdx.x;
  int cpe = H >> 2;  // chunks per edge
  if (idx >= E * cpe) return;
  int e = idx / cpe;
  int c = (idx % cpe) << 2;
  int s = ei[e];
  int d = ei[E + e];
  if ((unsigned)s >= (unsigned)N || (unsigned)d >= (unsigned)N) return;
  float w = dinv[s] * dinv[d];
  const float4 pv = *(const float4*)(p + (size_t)s * H + c);
  float* ap = agg + (size_t)d * H + c;
  atomicAdd(ap + 0, pv.x * w);
  atomicAdd(ap + 1, pv.y * w);
  atomicAdd(ap + 2, pv.z * w);
  atomicAdd(ap + 3, pv.w * w);
}

// ---------------- elementwise ----------------

__global__ void lrelu_k(float* __restrict__ x, int n) {
  int i = blockIdx.x * blockDim.x + threadIdx.x;
  if (i < n) {
    float v = x[i];
    x[i] = (v > 0.f) ? v : LSLOPE * v;
  }
}

// ---------------- WMMA GEMM ----------------
// C[M x Nout] = act( A[M x K](f32) * Bt^T + ... ), Bt is f16 [Npad][K] row-major:
// Bt[n*ldb + k] == B_math(k, n).

// A fragment: 16x32 f16, per ISA layout. lane: m=l&15, kh=l>>4;
// VGPR v holds K = (v/4)*16 + kh*8 + (v%4)*2, +1
__device__ __forceinline__ v16h load_a_frag(const float* __restrict__ A, int lda, int m0, int k0) {
  int l = threadIdx.x & 31;
  int m = m0 + (l & 15);
  int kh = l >> 4;
  union { v16h v; _Float16 e[16]; } u;
  const float* row = A + (size_t)m * lda + k0;
#pragma unroll
  for (int r = 0; r < 8; ++r) {
    int kb = (r >> 2) * 16 + kh * 8 + (r & 3) * 2;
    float2 f = *(const float2*)(row + kb);
    u.e[2 * r]     = (_Float16)f.x;
    u.e[2 * r + 1] = (_Float16)f.y;
  }
  return u.v;
}

// B fragment: 32x16 f16. lane: n=l&15, kh=l>>4; VGPR v holds K = (v/4)*16 + kh*8 + (v%4)*2, +1
__device__ __forceinline__ v16h load_b_frag(const _Float16* __restrict__ Bt, int ldb, int n0, int k0) {
  int l = threadIdx.x & 31;
  int n = n0 + (l & 15);
  int kh = l >> 4;
  union { v16h v; h2v p[8]; } u;
  const _Float16* row = Bt + (size_t)n * ldb + k0 + kh * 8;
#pragma unroll
  for (int r = 0; r < 8; ++r) {
    u.p[r] = *(const h2v*)(row + 2 * r + (r >> 2) * 8);
  }
  return u.v;
}

// epilogue: C/D layout: VGPR r -> M = m0 + r + 8*(lane>>4), N = n0base + jt*16 + (lane&15)
// MODE 3 reads the previous value from `prev` (same row-major layout/ld as C).
template <int NT, int MODE>
__device__ __forceinline__ void store_tile(const v8f* acc, const float* __restrict__ bias,
                                           const float* __restrict__ add,
                                           const float* __restrict__ prev,
                                           float* __restrict__ C,
                                           int m0, int n0base, int Nout, int ldc, float eps) {
  int l = threadIdx.x & 31;
  int nlo = l & 15;
  int kh = l >> 4;
#pragma unroll
  for (int jt = 0; jt < NT; ++jt) {
    int n = n0base + jt * 16 + nlo;
    if (n >= Nout) continue;
    float bv = (MODE == 1 || MODE == 2 || MODE == 3) ? bias[n] : 0.f;
#pragma unroll
    for (int r = 0; r < 8; ++r) {
      int m = m0 + r + 8 * kh;
      size_t idx = (size_t)m * ldc + n;
      float v = acc[jt][r];
      float outv;
      if (MODE == 0) {
        outv = v;
      } else if (MODE == 1) {
        float t = v + bv;
        outv = (t > 0.f) ? t : LSLOPE * t;
      } else if (MODE == 2) {
        outv = v + bv;
      } else {  // antisym update, out of place: C != A, prev = old h
        float oldv = prev[idx];
        float ag = add[idx];
        outv = oldv + eps * tanhf(v + ag + bv);
      }
      C[idx] = outv;
    }
  }
}

// MODE: 0 = plain, 1 = leaky_relu(acc+bias), 2 = acc+bias, 3 = antisym C = prev + eps*tanh(acc+add+bias)
// Each wave: TWO 16-row M tiles x NT column tiles (n-chunk from blockIdx.y).
template <int NT, int MODE>
__global__ void __launch_bounds__(128)
gemm_wmma_k(const float* __restrict__ A, const _Float16* __restrict__ Bt,
            const float* __restrict__ bias, const float* __restrict__ add,
            const float* __restrict__ prev, float* __restrict__ C,
            int Mtiles, int K, int lda, int ldb, int Nout, int ldc, float eps) {
  int pair = blockIdx.x * blockDim.y + threadIdx.y;
  int npairs = (Mtiles + 1) >> 1;
  if (pair >= npairs) return;
  int n0base = blockIdx.y * (NT * 16);
  int t0 = pair * 2;
  int t1 = t0 + 1;
  bool has1 = (t1 < Mtiles);
  int m0 = t0 * 16;
  int m1 = (has1 ? t1 : t0) * 16;  // clamp loads; stores guarded by has1

  v8f acc0[NT], acc1[NT];
#pragma unroll
  for (int jt = 0; jt < NT; ++jt) { acc0[jt] = (v8f)(0.f); acc1[jt] = (v8f)(0.f); }

  for (int k0 = 0; k0 < K; k0 += 32) {
    v16h a0 = load_a_frag(A, lda, m0, k0);
    v16h a1 = load_a_frag(A, lda, m1, k0);
#pragma unroll
    for (int jt = 0; jt < NT; ++jt) {
      v16h b = load_b_frag(Bt, ldb, n0base + jt * 16, k0);
      acc0[jt] = __builtin_amdgcn_wmma_f32_16x16x32_f16(
          false, a0, false, b, (short)0, acc0[jt], false, false);
      acc1[jt] = __builtin_amdgcn_wmma_f32_16x16x32_f16(
          false, a1, false, b, (short)0, acc1[jt], false, false);
    }
  }

  store_tile<NT, MODE>(acc0, bias, add, prev, C, m0, n0base, Nout, ldc, eps);
  if (has1) store_tile<NT, MODE>(acc1, bias, add, prev, C, m1, n0base, Nout, ldc, eps);
}

// ---------------- log_softmax over rows of 40, one wave32 per row ----------------

__global__ void log_softmax_k(float* __restrict__ out, int N, int C) {
  int row = blockIdx.x * blockDim.y + threadIdx.y;
  if (row >= N) return;
  int lane = threadIdx.x;  // 0..31
  float* r = out + (size_t)row * C;
  float v0 = r[lane];  // C=40, lanes 0..31 valid
  float v1 = (lane + 32 < C) ? r[lane + 32] : -INFINITY;
  float m = fmaxf(v0, v1);
#pragma unroll
  for (int o = 16; o > 0; o >>= 1) m = fmaxf(m, __shfl_xor(m, o, 32));
  float s = expf(v0 - m) + ((lane + 32 < C) ? expf(v1 - m) : 0.f);
#pragma unroll
  for (int o = 16; o > 0; o >>= 1) s += __shfl_xor(s, o, 32);
  float lse = m + logf(s);
  r[lane] = v0 - lse;
  if (lane + 32 < C) r[lane + 32] = v1 - lse;
}

// ---------------- host launch ----------------

extern "C" void kernel_launch(void* const* d_in, const int* in_sizes, int n_in,
                              void* d_out, int out_size, void* d_ws, size_t ws_size,
                              hipStream_t stream) {
  const float* x      = (const float*)d_in[0];
  const int*   ei     = (const int*)d_in[1];
  const float* w_hid  = (const float*)d_in[2];
  const float* b_hid  = (const float*)d_in[3];
  const float* W_a1   = (const float*)d_in[4];
  const float* gcn_w1 = (const float*)d_in[5];
  const float* b_a1   = (const float*)d_in[6];
  const float* w_hid2 = (const float*)d_in[7];
  const float* b_hid2 = (const float*)d_in[8];
  const float* W_a2   = (const float*)d_in[9];
  const float* gcn_w2 = (const float*)d_in[10];
  const float* b_a2   = (const float*)d_in[11];
  const float* w_fc   = (const float*)d_in[12];
  const float* b_fc   = (const float*)d_in[13];
  float* out = (float*)d_out;

  const int N = GN, E = GE, IN = GIN, H = GH, H2 = GH / 2, OUT = GOUT;
  const int Mtiles = N / 16;  // 3125

  // carve workspace
  char* base = (char*)d_ws;
  size_t off = 0;
  auto carve = [&](size_t bytes) -> char* {
    char* p = base + off;
    off = (off + bytes + 255) & ~(size_t)255;
    return p;
  };
  float* deg  = (float*)carve((size_t)N * 4);
  float* dinv = (float*)carve((size_t)N * 4);
  float* ha   = (float*)carve((size_t)N * H * 4);
  float* hb   = (float*)carve((size_t)N * H * 4);
  float* p    = (float*)carve((size_t)N * H * 4);   // reused as p2
  float* agg  = (float*)carve((size_t)N * H * 4);   // reused as agg2
  float* h2a  = (float*)carve((size_t)N * H2 * 4);
  float* h2b  = (float*)carve((size_t)N * H2 * 4);
  _Float16* bwhid  = (_Float16*)carve((size_t)H * IN * 2);
  _Float16* bgcn1  = (_Float16*)carve((size_t)H * H * 2);
  _Float16* baw1   = (_Float16*)carve((size_t)H * H * 2);
  _Float16* bw2    = (_Float16*)carve((size_t)H2 * H * 2);
  _Float16* bgcn2  = (_Float16*)carve((size_t)H2 * H2 * 2);
  _Float16* baw2   = (_Float16*)carve((size_t)H2 * H2 * 2);
  _Float16* bwfc   = (_Float16*)carve((size_t)48 * H2 * 2);
  (void)ws_size; (void)n_in; (void)in_sizes; (void)out_size;

  const int T = 256;
  auto g1 = [&](int n) { return dim3((n + T - 1) / T); };

  // prep: weight conversion
  cvt_copy_k<<<g1(H * IN), T, 0, stream>>>(w_hid, bwhid, H * IN);
  cvt_copy_k<<<g1(H2 * H), T, 0, stream>>>(w_hid2, bw2, H2 * H);
  cvt_transpose_k<<<g1(H * H), T, 0, stream>>>(gcn_w1, bgcn1, H, H);
  cvt_transpose_k<<<g1(H2 * H2), T, 0, stream>>>(gcn_w2, bgcn2, H2, H2);
  antisym_prep_k<<<g1(H * H), T, 0, stream>>>(W_a1, baw1, H, GAMMAV);
  antisym_prep_k<<<g1(H2 * H2), T, 0, stream>>>(W_a2, baw2, H2, GAMMAV);
  wfc_prep_k<<<g1(48 * H2), T, 0, stream>>>(w_fc, bwfc, H2, OUT, 48);

  // degree / norm
  deg_init_k<<<g1(N), T, 0, stream>>>(deg, N);
  deg_acc_k<<<g1(E), T, 0, stream>>>(ei, deg, E, N);
  dinv_k<<<g1(N), T, 0, stream>>>(deg, dinv, N);

  const int npairs = (Mtiles + 1) / 2;  // 1563
  dim3 gblock(32, 4);
  const int gx = (npairs + 3) / 4;      // 391

  // h = leaky_relu(x @ w_hid^T + b_hid)   [NT=4, 2 n-chunks]
  gemm_wmma_k<4, 1><<<dim3(gx, 2), gblock, 0, stream>>>(
      x, bwhid, b_hid, nullptr, nullptr, ha, Mtiles, IN, IN, IN, H, H, 0.f);

  // antisym conv 1: 3 iterations, ping-pong ha <-> hb
  float* hc = ha;
  float* hn = hb;
  for (int it = 0; it < 3; ++it) {
    gemm_wmma_k<4, 0><<<dim3(gx, 2), gblock, 0, stream>>>(
        hc, bgcn1, nullptr, nullptr, nullptr, p, Mtiles, H, H, H, H, H, 0.f);
    agg_init_k<<<g1(N * H), T, 0, stream>>>(p, dinv, agg, N, H);
    scatter_k<<<g1(E * (H / 4)), T, 0, stream>>>(ei, p, dinv, agg, E, H, N);
    gemm_wmma_k<4, 3><<<dim3(gx, 2), gblock, 0, stream>>>(
        hc, baw1, b_a1, agg, hc, hn, Mtiles, H, H, H, H, H, EPSV);
    float* tmp = hc; hc = hn; hn = tmp;
  }

  // h = leaky_relu(h); h2 = leaky_relu(h @ w_hid2^T + b_hid2)
  lrelu_k<<<g1(N * H), T, 0, stream>>>(hc, N * H);
  gemm_wmma_k<4, 1><<<dim3(gx, 1), gblock, 0, stream>>>(
      hc, bw2, b_hid2, nullptr, nullptr, h2a, Mtiles, H, H, H, H2, H2, 0.f);

  // antisym conv 2: 1 iteration (H2 = 64)
  gemm_wmma_k<4, 0><<<dim3(gx, 1), gblock, 0, stream>>>(
      h2a, bgcn2, nullptr, nullptr, nullptr, p, Mtiles, H2, H2, H2, H2, H2, 0.f);
  agg_init_k<<<g1(N * H2), T, 0, stream>>>(p, dinv, agg, N, H2);
  scatter_k<<<g1(E * (H2 / 4)), T, 0, stream>>>(ei, p, dinv, agg, E, H2, N);
  gemm_wmma_k<4, 3><<<dim3(gx, 1), gblock, 0, stream>>>(
      h2a, baw2, b_a2, agg, h2a, h2b, Mtiles, H2, H2, H2, H2, H2, EPSV);

  // logits = h2 @ w_fc^T + b_fc  (Nout=40, padded B to 48)
  gemm_wmma_k<3, 2><<<dim3(gx, 1), gblock, 0, stream>>>(
      h2b, bwfc, b_fc, nullptr, nullptr, out, Mtiles, H2, H2, H2, OUT, OUT, 0.f);

  // log_softmax rows
  dim3 sblock(32, 8);
  dim3 sgrid((N + 7) / 8);
  log_softmax_k<<<sgrid, sblock, 0, stream>>>(out, N, OUT);
}
